// MultiHeadAttention_8667244003893
// MI455X (gfx1250) — matrix operand order repaired and checked by reading, not compiled, not run on hardware
//
#include <hip/hip_runtime.h>
#include <hip/hip_bf16.h>

#define D_MODEL 1024
#define HEADS   16
#define SEQ     2048
#define BATCH   4
#define DHEAD   64

typedef __attribute__((ext_vector_type(16))) _Float16 v16h;
typedef __attribute__((ext_vector_type(8)))  _Float16 v8h;
typedef __attribute__((ext_vector_type(8)))  float    v8f;
typedef __attribute__((ext_vector_type(4)))  float    v4f;

__device__ __forceinline__ v8f wmma_f16(v16h a, v16h b, v8f c) {
    return __builtin_amdgcn_wmma_f32_16x16x32_f16(false, a, false, b, (short)0, c, false, false);
}

__device__ __forceinline__ v8f zero_v8f() {
    v8f z;
#pragma unroll
    for (int i = 0; i < 8; ++i) z[i] = 0.0f;
    return z;
}

__device__ __forceinline__ v16h combine_v8h(v8h lo, v8h hi) {
    v16h r;
#pragma unroll
    for (int i = 0; i < 8; ++i) { r[i] = lo[i]; r[8 + i] = hi[i]; }
    return r;
}

// A-fragment (16x32 f16): lane holds K = {2*koff..2*koff+15} (contiguous), koff=(lane>>4)*8
__device__ __forceinline__ v16h ldA_f16(const _Float16* __restrict__ rowk, int koff) {
    v8h lo = *(const v8h*)(rowk + 2 * koff);
    v8h hi = *(const v8h*)(rowk + 2 * koff + 8);
    return combine_v8h(lo, hi);
}
// B-fragment: lane holds 16 contiguous K of its column (p already offset by (lane>>4)*16)
__device__ __forceinline__ v16h ldB_f16(const _Float16* __restrict__ p) {
    return *(const v16h*)p;
}

// ---------------------------------------------------------------------------
// Kernel 0: fp32 -> f16 streaming convert (8 elements/thread)
// ---------------------------------------------------------------------------
__global__ __launch_bounds__(256) void cvt_f16_kernel(
    const float* __restrict__ src, _Float16* __restrict__ dst, int n)
{
    const int i = (blockIdx.x * 256 + threadIdx.x) * 8;
    if (i + 8 <= n) {
        v4f a = *(const v4f*)(src + i);
        v4f b = *(const v4f*)(src + i + 4);
        v8h r;
#pragma unroll
        for (int j = 0; j < 4; ++j) { r[j] = (_Float16)a[j]; r[4 + j] = (_Float16)b[j]; }
        *(v8h*)(dst + i) = r;
    }
}

// ---------------------------------------------------------------------------
// Kernel 1: Q/K/V projections  out[b,h,s,dd] = sum_k x[b,s,k] * W[h*64+dd,k]
// Block 256 thr = 8 waves as 2(M)x4(N); wave tile 64x64 -> block 128x256.
// Pure f16 operands; fp32 accumulation.
// ---------------------------------------------------------------------------
__global__ __launch_bounds__(256) void qkv_proj_kernel(
    const _Float16* __restrict__ xq, const _Float16* __restrict__ xk,
    const _Float16* __restrict__ xv,
    const _Float16* __restrict__ Wq, const _Float16* __restrict__ Wk,
    const _Float16* __restrict__ Wv,
    _Float16* __restrict__ qh, _Float16* __restrict__ kh, _Float16* __restrict__ vh)
{
    const int z = blockIdx.z;
    const int b = z & 3;
    const int which = z >> 2;                         // 0=q 1=k 2=v
    const _Float16* x = (which == 0) ? xq : (which == 1) ? xk : xv;
    const _Float16* W = (which == 0) ? Wq : (which == 1) ? Wk : Wv;
    _Float16* out     = (which == 0) ? qh : (which == 1) ? kh : vh;

    const int wave  = threadIdx.x >> 5;
    const int lane  = threadIdx.x & 31;
    const int lrow  = lane & 15;
    const int lhalf = lane >> 4;
    const int m0 = blockIdx.x * 128 + (wave >> 2) * 64;
    const int n0 = blockIdx.y * 256 + (wave & 3) * 64;

    const _Float16* xb = x + (size_t)b * SEQ * D_MODEL;

    const _Float16* arow[4];
    const _Float16* brow[4];
#pragma unroll
    for (int i = 0; i < 4; ++i) {
        arow[i] = xb + (size_t)(m0 + i * 16 + lrow) * D_MODEL;
        brow[i] = W  + (size_t)(n0 + i * 16 + lrow) * D_MODEL;
    }

    v8f acc[4][4];
#pragma unroll
    for (int mi = 0; mi < 4; ++mi)
#pragma unroll
        for (int ni = 0; ni < 4; ++ni) acc[mi][ni] = zero_v8f();

    const int koffA = lhalf * 8;
    const int koffB = lhalf * 16;
    for (int k = 0; k < D_MODEL; k += 32) {
        v16h a[4], bf[4];
#pragma unroll
        for (int mi = 0; mi < 4; ++mi) a[mi]  = ldA_f16(arow[mi] + k, koffA);
#pragma unroll
        for (int ni = 0; ni < 4; ++ni) bf[ni] = ldB_f16(brow[ni] + k + koffB);
#pragma unroll
        for (int mi = 0; mi < 4; ++mi)
#pragma unroll
            for (int ni = 0; ni < 4; ++ni)
                acc[mi][ni] = wmma_f16(a[mi], bf[ni], acc[mi][ni]);
    }

    // Store C-layout -> head-major f16 [B,H,S,64]
#pragma unroll
    for (int ni = 0; ni < 4; ++ni) {
        const int e  = n0 + ni * 16 + lrow;
        const int hh = e >> 6;
        const int dd = e & 63;
        _Float16* op = out + ((size_t)(b * HEADS + hh) * SEQ) * DHEAD + dd;
#pragma unroll
        for (int mi = 0; mi < 4; ++mi) {
            const int mb = m0 + mi * 16 + 8 * lhalf;
#pragma unroll
            for (int r = 0; r < 8; ++r)
                op[(size_t)(mb + r) * DHEAD] = (_Float16)acc[mi][ni][r];
        }
    }
}

// ---------------------------------------------------------------------------
// Kernel 2: flash attention per (b,h). Wave owns 16 query rows. V^T tile is
// staged cooperatively by the whole block (all waves consume the same keys);
// per-wave causal/pad bound applied with a wave-uniform guard (EXEC stays 1s).
// ---------------------------------------------------------------------------
#define VT_STRIDE 40   // padded row stride (halfs); multiple of 8 -> 16B aligned

__global__ __launch_bounds__(256) void attn_kernel(
    const _Float16* __restrict__ qh, const _Float16* __restrict__ kh,
    const _Float16* __restrict__ vh, const int* __restrict__ sen_len,
    _Float16* __restrict__ ctxo)
{
    __shared__ _Float16 vT[64 * VT_STRIDE];        // [dd][j], shared by block
    __shared__ _Float16 Pm[8 * 512];               // per-wave 16x32 P tile

    const int b = blockIdx.z, h = blockIdx.y;
    const int wave  = threadIdx.x >> 5;
    const int lane  = threadIdx.x & 31;
    const int lrow  = lane & 15;
    const int lhalf = lane >> 4;
    const int q0blk = blockIdx.x * 128;
    const int q0    = q0blk + wave * 16;

    const size_t head = ((size_t)b * HEADS + h) * (size_t)SEQ * DHEAD;
    const _Float16* Q = qh + head;
    const _Float16* K = kh + head;
    const _Float16* V = vh + head;
    _Float16* Pw = Pm + wave * 512;

    const int koffA = lhalf * 8;

    v16h aQ[2];
#pragma unroll
    for (int c = 0; c < 2; ++c)
        aQ[c] = ldA_f16(Q + (size_t)(q0 + lrow) * DHEAD + c * 32, koffA);

    v8f ctx[4];
#pragma unroll
    for (int t = 0; t < 4; ++t) ctx[t] = zero_v8f();
    float mrun[8], lrun[8];
#pragma unroll
    for (int r = 0; r < 8; ++r) { mrun[r] = -3.0e38f; lrun[r] = 0.0f; }

    const int slen = sen_len[b];
    const int nk_blk = (slen < q0blk + 128) ? slen : (q0blk + 128);  // block bound
    const int nk_w   = (slen < q0 + 16)     ? slen : (q0 + 16);      // wave bound
    const int ntiles = (nk_blk + 31) >> 5;                           // >= 1

    // cooperative staging mapping: thread -> (key row, 8-wide dd chunk)
    const int srow = threadIdx.x >> 3;     // 0..31
    const int sch  = threadIdx.x & 7;      // 0..7

    for (int kt = 0; kt < ntiles; ++kt) {
        const int jb = kt * 32;

        __syncthreads();                   // vT consumers of prev tile done
        {
            v8h vv = *(const v8h*)(V + (size_t)(jb + srow) * DHEAD + sch * 8);
#pragma unroll
            for (int i = 0; i < 8; ++i)
                vT[(sch * 8 + i) * VT_STRIDE + srow] = vv[i];
        }
        __syncthreads();                   // vT tile ready

        if (jb < nk_w) {                   // wave-uniform
            // Scores: S(16x32) = Q(16x64) @ K^T, K-frags contiguous from global
            v8f sc[2];
            sc[0] = zero_v8f(); sc[1] = zero_v8f();
#pragma unroll
            for (int jt = 0; jt < 2; ++jt) {
#pragma unroll
                for (int c = 0; c < 2; ++c) {
                    v16h bk = ldB_f16(K + (size_t)(jb + jt * 16 + lrow) * DHEAD +
                                      c * 32 + lhalf * 16);
                    sc[jt] = wmma_f16(aQ[c], bk, sc[jt]);
                }
            }

            // Mask + online softmax (C-layout: row = r + 8*lhalf, col = jt*16 + lrow)
            float p0[8], p1[8];
#pragma unroll
            for (int r = 0; r < 8; ++r) {
                const int qrow = q0 + r + 8 * lhalf;
                const int j0 = jb + lrow;
                const int j1 = jb + 16 + lrow;
                float s0 = sc[0][r] * 0.125f;
                float s1 = sc[1][r] * 0.125f;
                if (j0 > qrow || j0 >= slen) s0 = -1.0e9f;
                if (j1 > qrow || j1 >= slen) s1 = -1.0e9f;

                float mx = fmaxf(s0, s1);
                mx = fmaxf(mx, __shfl_xor(mx, 1));
                mx = fmaxf(mx, __shfl_xor(mx, 2));
                mx = fmaxf(mx, __shfl_xor(mx, 4));
                mx = fmaxf(mx, __shfl_xor(mx, 8));
                const float mnew  = fmaxf(mrun[r], mx);
                const float alpha = __expf(mrun[r] - mnew);
                p0[r] = __expf(s0 - mnew);
                p1[r] = __expf(s1 - mnew);
                float sum = p0[r] + p1[r];
                sum += __shfl_xor(sum, 1);
                sum += __shfl_xor(sum, 2);
                sum += __shfl_xor(sum, 4);
                sum += __shfl_xor(sum, 8);
                lrun[r] = lrun[r] * alpha + sum;
                mrun[r] = mnew;
#pragma unroll
                for (int t = 0; t < 4; ++t) ctx[t][r] *= alpha;
            }

            // C-layout -> A-layout via per-wave LDS tile
#pragma unroll
            for (int r = 0; r < 8; ++r) {
                const int m = r + 8 * lhalf;
                Pw[m * 32 + lrow]      = (_Float16)p0[r];
                Pw[m * 32 + 16 + lrow] = (_Float16)p1[r];
            }
            v16h aP = ldA_f16(Pw + lrow * 32, koffA);

            // ctx(16x64) += P(16x32) @ V(32x64); B-frags contiguous from vT
#pragma unroll
            for (int t = 0; t < 4; ++t) {
                v16h bv = *(const v16h*)(vT + (t * 16 + lrow) * VT_STRIDE + lhalf * 16);
                ctx[t] = wmma_f16(aP, bv, ctx[t]);
            }
        }
    }

    // Normalize and store ctx as f16 [B,S,D] (e = h*64+dd)
#pragma unroll
    for (int r = 0; r < 8; ++r) {
        const float inv = 1.0f / lrun[r];
        const int s = q0 + r + 8 * lhalf;
        _Float16* op = ctxo + ((size_t)b * SEQ + s) * D_MODEL + h * DHEAD + lrow;
#pragma unroll
        for (int t = 0; t < 4; ++t) op[t * 16] = (_Float16)(ctx[t][r] * inv);
    }
}

// ---------------------------------------------------------------------------
// Kernel 3: fused  out = LayerNorm(ctx @ Wo^T + residual)
// Block = 32 rows x full D; wave tile 32 x 128 (2 M-subtiles x 8 N-subtiles).
// ---------------------------------------------------------------------------
__global__ __launch_bounds__(256) void oproj_ln_kernel(
    const _Float16* __restrict__ ctx, const _Float16* __restrict__ Wo,
    const float* __restrict__ resid, const float* __restrict__ gamma,
    const float* __restrict__ beta, float* __restrict__ out)
{
    __shared__ float rs[32][8];
    __shared__ float rq[32][8];

    const int wave  = threadIdx.x >> 5;
    const int lane  = threadIdx.x & 31;
    const int lrow  = lane & 15;
    const int lhalf = lane >> 4;
    const int m0 = blockIdx.x * 32;          // row block in flattened B*S
    const int n0 = wave * 128;               // this wave's column slice

    const _Float16* arow[2];
#pragma unroll
    for (int mi = 0; mi < 2; ++mi)
        arow[mi] = ctx + (size_t)(m0 + mi * 16 + lrow) * D_MODEL;
    const _Float16* brow[8];
#pragma unroll
    for (int ni = 0; ni < 8; ++ni)
        brow[ni] = Wo + (size_t)(n0 + ni * 16 + lrow) * D_MODEL;

    v8f acc[2][8];
#pragma unroll
    for (int mi = 0; mi < 2; ++mi)
#pragma unroll
        for (int ni = 0; ni < 8; ++ni) acc[mi][ni] = zero_v8f();

    const int koffA = lhalf * 8;
    const int koffB = lhalf * 16;
    for (int k = 0; k < D_MODEL; k += 32) {
        v16h a[2], bf[8];
#pragma unroll
        for (int mi = 0; mi < 2; ++mi) a[mi]  = ldA_f16(arow[mi] + k, koffA);
#pragma unroll
        for (int ni = 0; ni < 8; ++ni) bf[ni] = ldB_f16(brow[ni] + k + koffB);
#pragma unroll
        for (int mi = 0; mi < 2; ++mi)
#pragma unroll
            for (int ni = 0; ni < 8; ++ni)
                acc[mi][ni] = wmma_f16(a[mi], bf[ni], acc[mi][ni]);
    }

    // Add residual; per-row partial stats over this wave's 128 columns
#pragma unroll
    for (int mi = 0; mi < 2; ++mi) {
#pragma unroll
        for (int r = 0; r < 8; ++r) {
            const int m = mi * 16 + r + 8 * lhalf;             // 0..31
            const size_t rowbase = (size_t)(m0 + m) * D_MODEL;
            float s = 0.0f, sq = 0.0f;
#pragma unroll
            for (int ni = 0; ni < 8; ++ni) {
                const int e = n0 + ni * 16 + lrow;
                float v = acc[mi][ni][r] + resid[rowbase + e];
                acc[mi][ni][r] = v;
                s += v; sq += v * v;
            }
            s  += __shfl_xor(s, 1);  sq += __shfl_xor(sq, 1);
            s  += __shfl_xor(s, 2);  sq += __shfl_xor(sq, 2);
            s  += __shfl_xor(s, 4);  sq += __shfl_xor(sq, 4);
            s  += __shfl_xor(s, 8);  sq += __shfl_xor(sq, 8);
            if (lrow == 0) { rs[m][wave] = s; rq[m][wave] = sq; }
        }
    }
    __syncthreads();

    // Load gamma/beta for this wave's columns
    float gv[8], bv[8];
#pragma unroll
    for (int ni = 0; ni < 8; ++ni) {
        const int e = n0 + ni * 16 + lrow;
        gv[ni] = gamma[e];
        bv[ni] = beta[e];
    }

    // Finalize LN per row and store
#pragma unroll
    for (int mi = 0; mi < 2; ++mi) {
#pragma unroll
        for (int r = 0; r < 8; ++r) {
            const int m = mi * 16 + r + 8 * lhalf;
            float S = 0.0f, Q = 0.0f;
#pragma unroll
            for (int w = 0; w < 8; ++w) { S += rs[m][w]; Q += rq[m][w]; }
            const float mean = S * (1.0f / D_MODEL);
            const float var  = Q * (1.0f / D_MODEL) - mean * mean;
            const float inv  = rsqrtf(var + 1e-5f);
            float* orow = out + (size_t)(m0 + m) * D_MODEL;
#pragma unroll
            for (int ni = 0; ni < 8; ++ni) {
                const int e = n0 + ni * 16 + lrow;
                orow[e] = (acc[mi][ni][r] - mean) * inv * gv[ni] + bv[ni];
            }
        }
    }
}

// ---------------------------------------------------------------------------
extern "C" void kernel_launch(void* const* d_in, const int* in_sizes, int n_in,
                              void* d_out, int out_size, void* d_ws, size_t ws_size,
                              hipStream_t stream) {
    (void)in_sizes; (void)n_in; (void)out_size; (void)ws_size;
    const float* q   = (const float*)d_in[0];
    const float* k   = (const float*)d_in[1];
    const float* v   = (const float*)d_in[2];
    const float* Wq  = (const float*)d_in[3];
    const float* Wk  = (const float*)d_in[4];
    const float* Wv  = (const float*)d_in[5];
    const float* Wo  = (const float*)d_in[6];
    const float* g   = (const float*)d_in[7];
    const float* bt  = (const float*)d_in[8];
    const int*   sl  = (const int*)d_in[9];
    float* out = (float*)d_out;

    const size_t nBSD = (size_t)BATCH * SEQ * D_MODEL;   // 8,388,608
    const size_t nDD  = (size_t)D_MODEL * D_MODEL;       // 1,048,576
    char* w = (char*)d_ws;
    _Float16* xqf = (_Float16*)w; w += nBSD * sizeof(_Float16);
    _Float16* xkf = (_Float16*)w; w += nBSD * sizeof(_Float16);
    _Float16* xvf = (_Float16*)w; w += nBSD * sizeof(_Float16);
    _Float16* Wqf = (_Float16*)w; w += nDD  * sizeof(_Float16);
    _Float16* Wkf = (_Float16*)w; w += nDD  * sizeof(_Float16);
    _Float16* Wvf = (_Float16*)w; w += nDD  * sizeof(_Float16);
    _Float16* Wof = (_Float16*)w; w += nDD  * sizeof(_Float16);
    _Float16* qhh = (_Float16*)w; w += nBSD * sizeof(_Float16);
    _Float16* khh = (_Float16*)w; w += nBSD * sizeof(_Float16);
    _Float16* vhh = (_Float16*)w; w += nBSD * sizeof(_Float16);
    _Float16* ctx = (_Float16*)w; w += nBSD * sizeof(_Float16);

    dim3 blk(256, 1, 1);
    const int gBSD = (int)(nBSD / (256 * 8));
    const int gDD  = (int)(nDD  / (256 * 8));
    cvt_f16_kernel<<<gBSD, blk, 0, stream>>>(q,  xqf, (int)nBSD);
    cvt_f16_kernel<<<gBSD, blk, 0, stream>>>(k,  xkf, (int)nBSD);
    cvt_f16_kernel<<<gBSD, blk, 0, stream>>>(v,  xvf, (int)nBSD);
    cvt_f16_kernel<<<gDD,  blk, 0, stream>>>(Wq, Wqf, (int)nDD);
    cvt_f16_kernel<<<gDD,  blk, 0, stream>>>(Wk, Wkf, (int)nDD);
    cvt_f16_kernel<<<gDD,  blk, 0, stream>>>(Wv, Wvf, (int)nDD);
    cvt_f16_kernel<<<gDD,  blk, 0, stream>>>(Wo, Wof, (int)nDD);

    qkv_proj_kernel<<<dim3(SEQ / 128, D_MODEL / 256, BATCH * 3), blk, 0, stream>>>(
        xqf, xkf, xvf, Wqf, Wkf, Wvf, qhh, khh, vhh);
    attn_kernel<<<dim3(SEQ / 128, HEADS, BATCH), blk, 0, stream>>>(qhh, khh, vhh, sl, ctx);
    oproj_ln_kernel<<<dim3((BATCH * SEQ) / 32, 1, 1), blk, 0, stream>>>(
        ctx, Wof, q, g, bt, out);
}
